// NMS3dAndComposeAAff_26405458936261
// MI455X (gfx1250) — matrix operand order repaired and checked by reading, not compiled, not run on hardware
//
#include <hip/hip_runtime.h>
#include <stdint.h>

#define H 2048
#define W 2048
#define NPIX (H * W)
#define KFEAT 4096
#define TILE 32
#define HALO 34

typedef __attribute__((ext_vector_type(16))) _Float16 v16h;
typedef __attribute__((ext_vector_type(8)))  float    v8f;
typedef __attribute__((ext_vector_type(4)))  unsigned int v4u;
typedef __attribute__((ext_vector_type(8)))  int      v8i_;
typedef __attribute__((ext_vector_type(4)))  int      v4i_;

// generic -> LDS(AS3) offset (32-bit LDS byte address for TDM descriptor)
__device__ __forceinline__ unsigned __lds_off(const void* p) {
    return (unsigned)(unsigned long long)(__attribute__((address_space(3))) const void*)p;
}

// ---------------- workspace layout (bytes) ----------------
static constexpr size_t WS_NMSED = 0;                         // float[NPIX]
static constexpr size_t WS_H1    = (size_t)NPIX * 4;          // u32[2048]
static constexpr size_t WS_H2    = WS_H1 + 2048 * 4;          // u32[2048]
static constexpr size_t WS_H3    = WS_H2 + 2048 * 4;          // u32[1024]
static constexpr size_t WS_SEL   = WS_H3 + 1024 * 4;          // u32[8]
static constexpr size_t WS_CVAL  = WS_SEL + 8 * 4;            // float[KFEAT]
static constexpr size_t WS_CIDX  = WS_CVAL + KFEAT * 4;       // u32[KFEAT]
#define ZERO_WORDS (2048 + 2048 + 1024 + 8)                   // u32s to zero (h1..sel)

// ---------------- init: zero histograms + counters ----------------
__global__ void k_init(unsigned int* p) {
    int i = blockIdx.x * blockDim.x + threadIdx.x;
    if (i < ZERO_WORDS) p[i] = 0u;
}

// ---------------- fused 3x3x3 NMS + masks + octave update ----------------
// Interior tiles: halo staged into LDS via Tensor Data Mover (3 x 2D tile descriptors),
// synchronized with s_wait_tensorcnt. Boundary tiles: VALU fill with -inf halo.
__global__ __launch_bounds__(256) void k_nms(const float* __restrict__ low,
                                             const float* __restrict__ cur,
                                             const float* __restrict__ high,
                                             const float* __restrict__ oct,
                                             float* __restrict__ nmsed,
                                             float* __restrict__ oct_out) {
    __shared__ float sm[3][HALO][HALO];  // contiguous: matches TDM tile layout
    const int tx = blockIdx.x * TILE, ty = blockIdx.y * TILE;
    const float NEG = -__builtin_inff();
    const bool interior = (tx >= 1) && (ty >= 1) && (tx + TILE + 1 <= W) && (ty + TILE + 1 <= H);

    if (interior) {
        if (threadIdx.x == 0) {  // TDM ignores EXEC; issue once (wave 0 only reaches here)
            const float* planes[3] = { low, cur, high };
            unsigned long long ga_off = ((unsigned long long)(ty - 1) * W + (tx - 1)) * 4ull;
            #pragma unroll
            for (int z = 0; z < 3; ++z) {
                unsigned long long ga = (unsigned long long)planes[z] + ga_off;
                unsigned lds = __lds_off(&sm[z][0][0]);
                // D# group0: [1:0]=count=1, [63:32]=lds_addr, [120:64]=global_addr, [127:126]=type=2
                v4u g0 = { 1u, lds, (unsigned)ga,
                           (unsigned)((ga >> 32) & 0x1FFFFFFu) | 0x80000000u };
                // D# group1: data_size=4B(2), tensor 2048x2048, tile 34x34, dim0_stride=2048
                v8i_ g1 = { (int)(2u << 16),                       // wg_mask=0, data_size=2
                            0,                                      // atomic barrier addr
                            (int)(((unsigned)W & 0xFFFFu) << 16),   // tensor_dim0[15:0] @ [95:80]
                            (int)(((unsigned)W >> 16) | (((unsigned)H & 0xFFFFu) << 16)), // dim0 hi, dim1 lo
                            (int)((34u << 16) | ((unsigned)H >> 16)), // dim1 hi, tile_dim0=34
                            (int)34,                                // tile_dim1=34, tile_dim2=0
                            (int)W,                                 // tensor_dim0_stride[31:0]
                            0 };                                    // stride hi / dim1_stride lo
                v4i_ g2 = { 0, 0, 0, 0 };
                v4i_ g3 = { 0, 0, 0, 0 };
                v8i_ g4 = { 0, 0, 0, 0, 0, 0, 0, 0 };
                __builtin_amdgcn_tensor_load_to_lds(g0, g1, g2, g3, g4, 0);
            }
        }
    } else {
        for (int l = threadIdx.x; l < HALO * HALO; l += 256) {
            int ly = l / HALO, lx = l % HALO;
            int gy = ty + ly - 1, gx = tx + lx - 1;
            bool in = (gy >= 0) & (gy < H) & (gx >= 0) & (gx < W);
            int gi = gy * W + gx;
            sm[0][ly][lx] = in ? low[gi]  : NEG;
            sm[1][ly][lx] = in ? cur[gi]  : NEG;
            sm[2][ly][lx] = in ? high[gi] : NEG;
        }
    }
    __builtin_amdgcn_s_wait_tensorcnt(0);
    __syncthreads();

    #pragma unroll
    for (int i = 0; i < (TILE * TILE) / 256; ++i) {
        int p = threadIdx.x + i * 256;
        int py = p / TILE, px = p % TILE;
        float mp = NEG;
        #pragma unroll
        for (int z = 0; z < 3; ++z)
            #pragma unroll
            for (int dy = 0; dy < 3; ++dy)
                #pragma unroll
                for (int dx = 0; dx < 3; ++dx)
                    mp = fmaxf(mp, sm[z][py + dy][px + dx]);
        float c = sm[1][py + 1][px + 1];
        int gy = ty + py, gx = tx + px;
        float v = ((c - mp + 1e-5f) > 0.f) ? c : 0.f;
        if (gy < 1 || gy >= H - 1 || gx < 1 || gx >= W - 1) v = 0.f;
        float o = oct[gy * W + gx];
        v *= (1.f - o);
        nmsed[gy * W + gx]   = v;
        oct_out[gy * W + gx] = (float)(unsigned char)(o + v);  // astype(uint8) truncation
    }
}

// ---------------- radix-select pass 1: bits[31:21] ----------------
__global__ __launch_bounds__(256) void k_hist1(const float* __restrict__ nm, unsigned int* __restrict__ h1) {
    __shared__ unsigned int lh[2048];
    for (int i = threadIdx.x; i < 2048; i += blockDim.x) lh[i] = 0u;
    __syncthreads();
    int stride = gridDim.x * blockDim.x;
    for (int i = blockIdx.x * blockDim.x + threadIdx.x; i < NPIX; i += stride) {
        __builtin_prefetch(&nm[i + stride], 0, 1);
        atomicAdd(&lh[__float_as_uint(nm[i]) >> 21], 1u);
    }
    __syncthreads();
    for (int i = threadIdx.x; i < 2048; i += blockDim.x)
        if (lh[i]) atomicAdd(&h1[i], lh[i]);
}

__global__ void k_scan1(const unsigned int* h1, unsigned int* sel) {
    if (threadIdx.x | blockIdx.x) return;
    unsigned int acc = 0;
    for (int b = 2047; b >= 0; --b) {
        unsigned int c = h1[b];
        if (acc + c >= KFEAT) { sel[0] = (unsigned int)b; sel[3] = acc; return; }
        acc += c;
    }
    sel[0] = 0u; sel[3] = acc;
}

// ---------------- pass 2: bits[20:10] within bin b1 ----------------
__global__ __launch_bounds__(256) void k_hist2(const float* __restrict__ nm, const unsigned int* __restrict__ sel,
                                               unsigned int* __restrict__ h2) {
    __shared__ unsigned int lh[2048];
    for (int i = threadIdx.x; i < 2048; i += blockDim.x) lh[i] = 0u;
    __syncthreads();
    unsigned int b1 = sel[0];
    int stride = gridDim.x * blockDim.x;
    for (int i = blockIdx.x * blockDim.x + threadIdx.x; i < NPIX; i += stride) {
        __builtin_prefetch(&nm[i + stride], 0, 1);
        unsigned int b = __float_as_uint(nm[i]);
        if ((b >> 21) == b1) atomicAdd(&lh[(b >> 10) & 0x7FFu], 1u);
    }
    __syncthreads();
    for (int i = threadIdx.x; i < 2048; i += blockDim.x)
        if (lh[i]) atomicAdd(&h2[i], lh[i]);
}

__global__ void k_scan2(const unsigned int* h2, unsigned int* sel) {
    if (threadIdx.x | blockIdx.x) return;
    unsigned int acc = sel[3];
    for (int b = 2047; b >= 0; --b) {
        unsigned int c = h2[b];
        if (acc + c >= KFEAT) { sel[1] = (unsigned int)b; sel[3] = acc; return; }
        acc += c;
    }
    sel[1] = 0u; sel[3] = acc;
}

// ---------------- pass 3: bits[9:0] within (b1,b2) ----------------
__global__ __launch_bounds__(256) void k_hist3(const float* __restrict__ nm, const unsigned int* __restrict__ sel,
                                               unsigned int* __restrict__ h3) {
    __shared__ unsigned int lh[1024];
    for (int i = threadIdx.x; i < 1024; i += blockDim.x) lh[i] = 0u;
    __syncthreads();
    unsigned int b1 = sel[0], b2 = sel[1];
    int stride = gridDim.x * blockDim.x;
    for (int i = blockIdx.x * blockDim.x + threadIdx.x; i < NPIX; i += stride) {
        __builtin_prefetch(&nm[i + stride], 0, 1);
        unsigned int b = __float_as_uint(nm[i]);
        if (((b >> 21) == b1) && (((b >> 10) & 0x7FFu) == b2)) atomicAdd(&lh[b & 0x3FFu], 1u);
    }
    __syncthreads();
    for (int i = threadIdx.x; i < 1024; i += blockDim.x)
        if (lh[i]) atomicAdd(&h3[i], lh[i]);
}

__global__ void k_scan3(const unsigned int* h3, unsigned int* sel) {
    if (threadIdx.x | blockIdx.x) return;
    unsigned int acc = sel[3];
    unsigned int b3 = 0;
    for (int b = 1023; b >= 0; --b) {
        unsigned int c = h3[b];
        if (acc + c >= KFEAT) { b3 = (unsigned int)b; break; }
        acc += c;
    }
    sel[2] = (sel[0] << 21) | (sel[1] << 10) | b3;  // threshold bit pattern T
    sel[3] = acc;                                   // count strictly above T
    sel[4] = KFEAT - acc;                           // how many ties at T to take
}

// ---------------- compaction: exactly KFEAT (value,index) pairs ----------------
__global__ __launch_bounds__(256) void k_compact(const float* __restrict__ nm, unsigned int* __restrict__ sel,
                                                 float* __restrict__ cval, unsigned int* __restrict__ cidx) {
    unsigned int T = sel[2], above = sel[3], needed = sel[4];
    int stride = gridDim.x * blockDim.x;
    for (int i = blockIdx.x * blockDim.x + threadIdx.x; i < NPIX; i += stride) {
        float v = nm[i];
        unsigned int b = __float_as_uint(v);
        if (b > T) {
            unsigned int p = atomicAdd(&sel[5], 1u);
            cval[p] = v; cidx[p] = (unsigned int)i;
        } else if (b == T) {
            unsigned int q = atomicAdd(&sel[6], 1u);
            if (q < needed) { cval[above + q] = v; cidx[above + q] = (unsigned int)i; }
        }
    }
}

// ---------------- single-WG bitonic sort of 4096 pairs (desc val, asc idx) ----------------
__global__ __launch_bounds__(1024) void k_sort(float* __restrict__ cval, unsigned int* __restrict__ cidx) {
    __shared__ float sv[KFEAT];
    __shared__ unsigned int si[KFEAT];
    for (int i = threadIdx.x; i < KFEAT; i += 1024) { sv[i] = cval[i]; si[i] = cidx[i]; }
    __syncthreads();
    for (int k = 2; k <= KFEAT; k <<= 1) {
        for (int j = k >> 1; j > 0; j >>= 1) {
            for (int t = threadIdx.x; t < KFEAT; t += 1024) {
                int ixj = t ^ j;
                if (ixj > t) {
                    float va = sv[t], vb = sv[ixj];
                    unsigned int ia = si[t], ib = si[ixj];
                    bool before = (va > vb) || ((va == vb) && (ia < ib)); // a precedes b
                    bool dirUp = ((t & k) == 0);
                    bool dosw = dirUp ? !before : before;
                    if (dosw) { sv[t] = vb; sv[ixj] = va; si[t] = ib; si[ixj] = ia; }
                }
            }
            __syncthreads();
        }
    }
    for (int i = threadIdx.x; i < KFEAT; i += 1024) { cval[i] = sv[i]; cidx[i] = si[i]; }
}

// ---------------- per-keypoint centroid conv via WMMA + LAF compose ----------------
// A (16x32 f16): rows 0..3 = coordinate weights over the 27 taps (z-0.5, y-0.5, x-0.5, 1)
// B (32x16 f16): 27 taps (K) x 16 keypoints (N). One v_wmma per 16 keypoints.
__global__ __launch_bounds__(128) void k_lafs(const float* __restrict__ low,
                                              const float* __restrict__ cur,
                                              const float* __restrict__ high,
                                              const float* __restrict__ aff,
                                              const float* __restrict__ cval,
                                              const unsigned int* __restrict__ cidx,
                                              float* __restrict__ vals,
                                              float* __restrict__ lafs) {
    int wave = (blockIdx.x * 128 + threadIdx.x) >> 5;  // 0..255
    int lane = threadIdx.x & 31;
    int nn = lane & 15;   // A row M / B column N
    int hi = lane >> 4;   // lane-group: K-range selector
    int kp = wave * 16 + nn;
    unsigned int idx = cidx[kp];
    int y = (int)(idx >> 11);      // /2048
    int x = (int)(idx & 2047u);

    // A-matrix per CDNA5 16-bit A 16x32 layout:
    //   lanes 0-15: VGPR0-3 -> K=0..7, VGPR4-7 -> K=16..23 ; lanes 16-31: +8
    v16h a;
    #pragma unroll
    for (int h2 = 0; h2 < 16; ++h2) {
        int v = h2 >> 1;
        int kk = ((v < 4) ? (2 * v) : (16 + 2 * (v - 4))) + (h2 & 1) + hi * 8;
        float wgt = 0.f;
        if (kk < 27) {
            int dz = kk / 9, r = kk % 9, dy = r / 3, dx = r % 3;
            if      (nn == 0) wgt = (float)dz - 0.5f;   // z-coordinate weight
            else if (nn == 1) wgt = (float)dy - 0.5f;   // y
            else if (nn == 2) wgt = (float)dx - 0.5f;   // x
            else if (nn == 3) wgt = 1.f;                // mass
        }
        a[h2] = (_Float16)wgt;
    }

    // B-matrix: lanes 0-15: elems 0-7 -> K=0..7, elems 8-15 -> K=16..23 ; lanes 16-31: +8; column N = lane&15
    v16h b;
    #pragma unroll
    for (int h2 = 0; h2 < 16; ++h2) {
        int kk = (h2 < 8) ? (hi * 8 + h2) : (16 + hi * 8 + (h2 - 8));
        float tv = 0.f;
        if (kk < 27) {
            int dz = kk / 9, r = kk % 9;
            int dy = r / 3 - 1, dx = r % 3 - 1;
            int yy = min(max(y + dy, 0), H - 1);
            int xx = min(max(x + dx, 0), W - 1);
            const float* pl = (dz == 0) ? low : ((dz == 1) ? cur : high);
            tv = pl[yy * W + xx];
        }
        b[h2] = (_Float16)tv;
    }

    v8f c = {};
    c = __builtin_amdgcn_wmma_f32_16x16x32_f16(false, a, false, b, (short)0, c, false, false);

    // C/D layout: lane n (0-15), VGPR m -> D[M=m][N=n]; our channels live in rows 0..3
    if (hi == 0) {
        float r = __builtin_amdgcn_rcpf(c[3] + 1e-8f);  // v_rcp_f32, no IEEE div expansion
        float s  = c[0] * r;
        float oy = c[1] * r + (float)y;
        float ox = c[2] * r + (float)x;
        float sn = s  * (1.f / 2048.f);  // / min(H,W)
        float yn = oy * (1.f / 2048.f);  // / H
        float xn = ox * (1.f / 2048.f);  // / W
        float a0 = aff[0 * NPIX + idx], a1 = aff[1 * NPIX + idx];
        float a2 = aff[2 * NPIX + idx], a3 = aff[3 * NPIX + idx];
        float* L = lafs + (size_t)kp * 6;
        L[0] = sn * a0; L[1] = sn * a1; L[2] = xn;
        L[3] = sn * a2; L[4] = sn * a3; L[5] = yn;
        vals[kp] = cval[kp];
    }
}

// ---------------- launch ----------------
extern "C" void kernel_launch(void* const* d_in, const int* in_sizes, int n_in,
                              void* d_out, int out_size, void* d_ws, size_t ws_size,
                              hipStream_t stream) {
    const float* low  = (const float*)d_in[0];
    const float* cur  = (const float*)d_in[1];
    const float* high = (const float*)d_in[2];
    const float* aff  = (const float*)d_in[3];
    const float* oct  = (const float*)d_in[4];
    // d_in[5] = num_features (fixed 4096 in this workload)

    float* vals    = (float*)d_out;              // [4096]
    float* lafs    = vals + KFEAT;               // [4096*6]
    float* oct_out = lafs + (size_t)KFEAT * 6;   // [H*W]

    char* ws = (char*)d_ws;
    float*        nmsed = (float*)(ws + WS_NMSED);
    unsigned int* h1    = (unsigned int*)(ws + WS_H1);
    unsigned int* h2    = (unsigned int*)(ws + WS_H2);
    unsigned int* h3    = (unsigned int*)(ws + WS_H3);
    unsigned int* sel   = (unsigned int*)(ws + WS_SEL);
    float*        cvalv = (float*)(ws + WS_CVAL);
    unsigned int* cidxv = (unsigned int*)(ws + WS_CIDX);

    k_init<<<(ZERO_WORDS + 255) / 256, 256, 0, stream>>>(h1);  // h1/h2/h3/sel contiguous
    dim3 g(W / TILE, H / TILE);
    k_nms<<<g, 256, 0, stream>>>(low, cur, high, oct, nmsed, oct_out);
    k_hist1<<<512, 256, 0, stream>>>(nmsed, h1);
    k_scan1<<<1, 1, 0, stream>>>(h1, sel);
    k_hist2<<<512, 256, 0, stream>>>(nmsed, sel, h2);
    k_scan2<<<1, 1, 0, stream>>>(h2, sel);
    k_hist3<<<512, 256, 0, stream>>>(nmsed, sel, h3);
    k_scan3<<<1, 1, 0, stream>>>(h3, sel);
    k_compact<<<512, 256, 0, stream>>>(nmsed, sel, cvalv, cidxv);
    k_sort<<<1, 1024, 0, stream>>>(cvalv, cidxv);
    k_lafs<<<64, 128, 0, stream>>>(low, cur, high, aff, cvalv, cidxv, vals, lafs);
}
